// GNNRanker_49787260895342
// MI455X (gfx1250) — matrix operand order repaired and checked by reading, not compiled, not run on hardware
//
#include <hip/hip_runtime.h>
#include <hip/hip_bf16.h>

#define USE_TDM 1           // TDM (tensor_load_to_lds) staging of B tile

#if defined(__has_builtin)
#  if __has_builtin(__builtin_amdgcn_sched_barrier)
#    define SCHED_FENCE() __builtin_amdgcn_sched_barrier(0)
#  endif
#endif
#ifndef SCHED_FENCE
#  define SCHED_FENCE() ((void)0)
#endif

// ---------------- problem constants ----------------
#define NN   50000          // nodes
#define NE   400000         // edges (before self loops)
#define ET   (NE + NN)      // edges + self loops
#define NX   389            // node feature dim
#define NQ   768            // question dim
#define KIN  (NX + NQ)      // 1157
#define KP   1184           // 1157 padded to multiple of 32
#define HID  128
#define HD   4              // heads
#define HC   (HD * HID)     // 512
#define NEG_SLOPE 0.2f
#define MT   5              // M sub-tiles per wave (80 rows per block)

// ---------------- WMMA types (gfx1250, wave32) ----------------
typedef __attribute__((ext_vector_type(16))) __bf16          v16bf;
typedef __attribute__((ext_vector_type(16))) unsigned short  v16us;
typedef __attribute__((ext_vector_type(8)))  float           v8f;
typedef __attribute__((ext_vector_type(4)))  unsigned        v4u;
typedef __attribute__((ext_vector_type(8)))  unsigned        v8u;

// f32 -> bf16 (round to nearest even), bit-level
__device__ __forceinline__ unsigned short f2bf(float f) {
    unsigned u = __float_as_uint(f);
    u += 0x7FFFu + ((u >> 16) & 1u);
    return (unsigned short)(u >> 16);
}

// order-preserving float<->uint transform for atomicMax on floats
__device__ __forceinline__ unsigned f_ord(float f) {
    unsigned u = __float_as_uint(f);
    return (u & 0x80000000u) ? ~u : (u | 0x80000000u);
}
__device__ __forceinline__ float ord_f(unsigned u) {
    return (u & 0x80000000u) ? __uint_as_float(u & 0x7FFFFFFFu)
                             : __uint_as_float(~u);
}
#define ORD_NEG_INF 0x007FFFFFu   // f_ord(-inf)

// Load a 16-half fragment from two contiguous 16-byte chunks
// (works for global or LDS pointers; LDS emits ds_load_b128)
__device__ __forceinline__ v16bf load_frag(const unsigned short* p0,
                                           const unsigned short* p1) {
    union { v16us us; uint4 q[2]; } u;
    u.q[0] = *(const uint4*)p0;
    u.q[1] = *(const uint4*)p1;
    return __builtin_bit_cast(v16bf, u.us);
}

#define WMMA_BF16(A_, B_, C_)                                                  \
    __builtin_amdgcn_wmma_f32_16x16x32_bf16(false, A_, false, B_, (short)0,    \
                                            C_, false, false)

// ============================================================
// Prep kernels
// ============================================================
__global__ void k_build_comb(const float* __restrict__ x,
                             const float* __restrict__ q,
                             const int*   __restrict__ bidx,
                             unsigned short* __restrict__ comb) {
    long long i = (long long)blockIdx.x * blockDim.x + threadIdx.x;
    if (i >= (long long)NN * KP) return;
    int n = (int)(i / KP);
    int c = (int)(i - (long long)n * KP);
    float v = 0.f;
    if (c < NX)       v = x[(long long)n * NX + c];
    else if (c < KIN) v = q[(long long)bidx[n] * NQ + (c - NX)];
    comb[i] = f2bf(v);
}

__global__ void k_conv_win(const float* __restrict__ W,
                           unsigned short* __restrict__ Wb) {
    int i = blockIdx.x * blockDim.x + threadIdx.x;
    if (i >= HID * KP) return;
    int r = i / KP, c = i - r * KP;
    float v = (c < KIN) ? W[r * KIN + c] : 0.f;
    Wb[i] = f2bf(v);
}

__global__ void k_conv_wl(const float* __restrict__ W,
                          unsigned short* __restrict__ Wb, int n) {
    int i = blockIdx.x * blockDim.x + threadIdx.x;
    if (i < n) Wb[i] = f2bf(W[i]);
}

// ============================================================
// GEMM 1: h = relu(comb[M,KP] @ Win[HID,KP]^T + b_in) -> bf16 [M,128]
// block = 256 = 8 waves; wave w owns N-tile n0=w*16 and MT M-subtiles.
// Per K-step: all 12 fragment loads are fenced (sched_barrier) before
// the 5 wmmas so loads issue as one clause with staggered waits, while
// named accumulators + unroll(1) stay pinned in place.
// ============================================================
__global__ void __launch_bounds__(256, 2)
k_proj_gemm(const unsigned short* __restrict__ A,   // [NN, KP] bf16
            const unsigned short* __restrict__ B,   // [HID, KP] bf16
            const float* __restrict__ bias,         // [HID]
            unsigned short* __restrict__ Hb) {      // [NN, HID] bf16
    const int lane = threadIdx.x & 31;
    const int wave = threadIdx.x >> 5;
    const int lo = lane & 15, hi = lane >> 4;
    const int m0 = blockIdx.x * (16 * MT);
    const int n0 = wave * 16;

    const size_t boff0 = (size_t)(n0 + lo) * KP + hi * 8;
    const size_t aoff0 = (size_t)(m0 + lo) * KP + hi * 8;
    const size_t S = (size_t)16 * KP;                // M-subtile stride

    v8f c0 = {}, c1 = {}, c2 = {}, c3 = {}, c4 = {};
#pragma unroll 1
    for (int k0 = 0; k0 < KP; k0 += 32) {
        const unsigned short* bp = B + (boff0 + k0);
        const unsigned short* ap = A + (aoff0 + k0);
        // ---- issue all loads up front ----
        v16bf b  = load_frag(bp,         bp + 16);
        v16bf a0 = load_frag(ap,         ap + 16);
        v16bf a1 = load_frag(ap + S,     ap + S + 16);
        v16bf a2 = load_frag(ap + 2 * S, ap + 2 * S + 16);
        v16bf a3 = load_frag(ap + 3 * S, ap + 3 * S + 16);
        v16bf a4 = load_frag(ap + 4 * S, ap + 4 * S + 16);
        SCHED_FENCE();                 // keep loads above the wmma burst
        // ---- then the matrix ops ----
        c0 = WMMA_BF16(a0, b, c0);
        c1 = WMMA_BF16(a1, b, c1);
        c2 = WMMA_BF16(a2, b, c2);
        c3 = WMMA_BF16(a3, b, c3);
        c4 = WMMA_BF16(a4, b, c4);
    }
    const int n = n0 + lo;
    const float bn = bias[n];
    v8f accs[MT] = {c0, c1, c2, c3, c4};
#pragma unroll
    for (int t = 0; t < MT; ++t) {
        const int mb = m0 + t * 16 + hi * 8;
#pragma unroll
        for (int r = 0; r < 8; ++r) {
            float v = accs[t][r] + bn;
            v = v > 0.f ? v : 0.f;
            Hb[(size_t)(mb + r) * HID + n] = f2bf(v);
        }
    }
}

// ============================================================
// GEMM 2: xh = h[M,128] @ Wl[512,128]^T -> f32 [M,512]
// B tile (128 N-rows x 128 K, 32KB) staged in LDS once per block via the
// Tensor Data Mover (TENSORcnt), fragments read with ds_load_b128.
// grid = (625, 4)
// ============================================================
__global__ void __launch_bounds__(256, 2)
k_layer_gemm(const unsigned short* __restrict__ A,   // [NN, 128] bf16
             const unsigned short* __restrict__ B,   // [512, 128] bf16
             float* __restrict__ XH) {               // [NN, 512] f32
    __shared__ unsigned short Bs[128 * HID];         // 32 KB

    const int lane = threadIdx.x & 31;
    const int wave = threadIdx.x >> 5;
    const int lo = lane & 15, hi = lane >> 4;

#if USE_TDM
    // ---- TDM: DMA the 128x128 bf16 tile at row blockIdx.y*128 into LDS ----
    if (wave == 0) {
        unsigned long long gaddr =
            (unsigned long long)(B + (size_t)blockIdx.y * 128 * HID);
        unsigned ldsaddr = (unsigned)(unsigned long long)(&Bs[0]); // LDS offset
        // D# group 0: count=1 | lds_addr | global_addr | type=2
        v4u g0;
        g0.x = 1u;                                   // count=1, user mode
        g0.y = ldsaddr;                              // lds_addr [63:32]
        g0.z = (unsigned)gaddr;                      // global_addr [95:64]
        g0.w = (unsigned)(gaddr >> 32) | (2u << 30); // addr[56:32] | type=2
        // D# group 1: data_size=2B, tensor 128x512, tile 128x128, stride 128
        v8u g1;
        g1.s0 = 1u << 16;          // workgroup_mask=0, data_size=1 (2 bytes)
        g1.s1 = 128u << 16;        // tensor_dim0 lo16 (=128)
        g1.s2 = 512u << 16;        // tensor_dim0 hi=0 | tensor_dim1 lo (=512)
        g1.s3 = 128u << 16;        // tensor_dim1 hi=0 | tile_dim0 (=128)
        g1.s4 = 128u;              // tile_dim1 (=128) | tile_dim2=0
        g1.s5 = 128u;              // tensor_dim0_stride lo (=128)
        g1.s6 = 0u;                // dim0_stride hi | dim1_stride lo
        g1.s7 = 0u;
        asm volatile("tensor_load_to_lds %0, %1"
                     :: "s"(g0), "s"(g1) : "memory");
        __builtin_amdgcn_s_wait_tensorcnt(0);
    }
    __syncthreads();
#else
    const unsigned short* Bg = B + (size_t)blockIdx.y * 128 * HID;
    for (int j = threadIdx.x; j < 128 * HID / 8; j += 256)
        *(uint4*)&Bs[j * 8] = *(const uint4*)&Bg[j * 8];
    __syncthreads();
#endif

    const int m0 = blockIdx.x * (16 * MT);
    const int n0 = wave * 16;
    const size_t aoff0 = (size_t)(m0 + lo) * HID + hi * 8;
    const unsigned short* bl0 = Bs + (size_t)(n0 + lo) * HID + hi * 8;
    const size_t S = (size_t)16 * HID;               // M-subtile stride

    v8f c0 = {}, c1 = {}, c2 = {}, c3 = {}, c4 = {};
#pragma unroll 1
    for (int k0 = 0; k0 < HID; k0 += 32) {
        const unsigned short* bp = bl0 + k0;
        const unsigned short* ap = A + (aoff0 + k0);
        // ---- issue all loads up front (B from LDS, A from global) ----
        v16bf b  = load_frag(bp,         bp + 16);   // ds_load_b128 x2
        v16bf a0 = load_frag(ap,         ap + 16);
        v16bf a1 = load_frag(ap + S,     ap + S + 16);
        v16bf a2 = load_frag(ap + 2 * S, ap + 2 * S + 16);
        v16bf a3 = load_frag(ap + 3 * S, ap + 3 * S + 16);
        v16bf a4 = load_frag(ap + 4 * S, ap + 4 * S + 16);
        SCHED_FENCE();                 // keep loads above the wmma burst
        // ---- then the matrix ops ----
        c0 = WMMA_BF16(a0, b, c0);
        c1 = WMMA_BF16(a1, b, c1);
        c2 = WMMA_BF16(a2, b, c2);
        c3 = WMMA_BF16(a3, b, c3);
        c4 = WMMA_BF16(a4, b, c4);
    }
    const int n = blockIdx.y * 128 + n0 + lo;
    v8f accs[MT] = {c0, c1, c2, c3, c4};
#pragma unroll
    for (int t = 0; t < MT; ++t) {
        const int mb = m0 + t * 16 + hi * 8;
#pragma unroll
        for (int r = 0; r < 8; ++r)
            XH[(size_t)(mb + r) * HC + n] = accs[t][r];
    }
}

// ============================================================
// attention logits: al_s[n,h] = dot(xh[n,h,:], a_src[h,:]) ; same for dst
// ============================================================
__global__ void k_attn_logits(const float* __restrict__ XH,
                              const float* __restrict__ as_,
                              const float* __restrict__ ad_,
                              float* __restrict__ als,
                              float* __restrict__ ald) {
    int i = blockIdx.x * blockDim.x + threadIdx.x;   // [0, NN*HD)
    if (i >= NN * HD) return;
    int h = i & 3, n = i >> 2;
    const float4* xr = (const float4*)(XH + (size_t)n * HC + h * HID);
    const float4* s  = (const float4*)(as_ + h * HID);
    const float4* d  = (const float4*)(ad_ + h * HID);
    float accs = 0.f, accd = 0.f;
#pragma unroll 4
    for (int c = 0; c < HID / 4; ++c) {
        float4 v = xr[c], sv = s[c], dv = d[c];
        accs += v.x * sv.x + v.y * sv.y + v.z * sv.z + v.w * sv.w;
        accd += v.x * dv.x + v.y * dv.y + v.z * dv.z + v.w * dv.w;
    }
    als[i] = accs;
    ald[i] = accd;
}

// init per-node segment state
__global__ void k_init_ms(unsigned* __restrict__ mmax, float* __restrict__ ssum) {
    int i = blockIdx.x * blockDim.x + threadIdx.x;
    if (i >= NN * HD) return;
    mmax[i] = ORD_NEG_INF;
    ssum[i] = 0.f;
}

__global__ void k_zero4(float4* __restrict__ p, long long n4) {
    long long i = (long long)blockIdx.x * blockDim.x + threadIdx.x;
    if (i < n4) p[i] = make_float4(0.f, 0.f, 0.f, 0.f);
}

__device__ __forceinline__ void edge_sd(int e, const int* ei, int& src, int& dst) {
    if (e < NE) { src = ei[e]; dst = ei[NE + e]; }
    else        { src = dst = e - NE; }   // self loop
}

// pass 1: e = leaky_relu(als[src]+ald[dst]); store; segment max over dst
__global__ void k_edge_max(const int* __restrict__ ei,
                           const float* __restrict__ als,
                           const float* __restrict__ ald,
                           float* __restrict__ elog,
                           unsigned* __restrict__ mmax) {
    int i = blockIdx.x * blockDim.x + threadIdx.x;   // [0, ET*HD)
    if (i >= ET * HD) return;
    int h = i & 3, e = i >> 2;
    int src, dst; edge_sd(e, ei, src, dst);
    float t = als[src * HD + h] + ald[dst * HD + h];
    t = t > 0.f ? t : NEG_SLOPE * t;
    elog[i] = t;
    atomicMax(&mmax[dst * HD + h], f_ord(t));
}

// pass 2: w = exp(e - m[dst]); store; segment sum over dst
__global__ void k_edge_exp(const int* __restrict__ ei,
                           float* __restrict__ elog,
                           const unsigned* __restrict__ mmax,
                           float* __restrict__ ssum) {
    int i = blockIdx.x * blockDim.x + threadIdx.x;
    if (i >= ET * HD) return;
    int h = i & 3, e = i >> 2;
    int src, dst; edge_sd(e, ei, src, dst);
    float w = __expf(elog[i] - ord_f(mmax[dst * HD + h]));
    elog[i] = w;
    atomicAdd(&ssum[dst * HD + h], w);
}

// pass 3: out[dst,h,c] += xh[src,h,c] * (w / (s[dst,h] + 1e-16))
__global__ void k_edge_scatter(const int* __restrict__ ei,
                               const float* __restrict__ elog,
                               const float* __restrict__ ssum,
                               const float* __restrict__ XH,
                               float* __restrict__ oacc) {
    long long i = (long long)blockIdx.x * blockDim.x + threadIdx.x; // [0, ET*128)
    if (i >= (long long)ET * HID) return;
    int c = (int)(i & (HID - 1));
    int e = (int)(i >> 7);
    int src, dst; edge_sd(e, ei, src, dst);
    const float* xr = XH + (size_t)src * HC + c;
    float* orow = oacc + (size_t)dst * HC + c;
#pragma unroll
    for (int h = 0; h < HD; ++h) {
        float w = elog[e * HD + h];
        float s = ssum[dst * HD + h];
        float alpha = w / (s + 1e-16f);
        atomicAdd(orow + h * HID, xr[h * HID] * alpha);
    }
}

// head mean + bias + relu -> next-layer h (both f32 and bf16)
__global__ void k_reduce(const float* __restrict__ oacc,
                         const float* __restrict__ bias,
                         float* __restrict__ Hf,
                         unsigned short* __restrict__ Hb) {
    int i = blockIdx.x * blockDim.x + threadIdx.x;   // [0, NN*HID)
    if (i >= NN * HID) return;
    int c = i & (HID - 1);
    int n = i >> 7;
    const float* o = oacc + (size_t)n * HC + c;
    float v = 0.25f * (o[0] + o[HID] + o[2 * HID] + o[3 * HID]) + bias[c];
    v = v > 0.f ? v : 0.f;
    Hf[i] = v;
    Hb[i] = f2bf(v);
}

// classifier + sigmoid
__global__ void k_logits(const float* __restrict__ Hf,
                         const float* __restrict__ Wc,
                         const float* __restrict__ bc,
                         float* __restrict__ out) {
    int n = blockIdx.x * blockDim.x + threadIdx.x;
    if (n >= NN) return;
    const float4* hr = (const float4*)(Hf + (size_t)n * HID);
    const float4* w4 = (const float4*)Wc;
    float acc = bc[0];
#pragma unroll 4
    for (int c = 0; c < HID / 4; ++c) {
        float4 h = hr[c], w = w4[c];
        acc += h.x * w.x + h.y * w.y + h.z * w.z + h.w * w.w;
    }
    out[n] = 1.f / (1.f + __expf(-acc));
}

// ============================================================
extern "C" void kernel_launch(void* const* d_in, const int* in_sizes, int n_in,
                              void* d_out, int out_size, void* d_ws, size_t ws_size,
                              hipStream_t stream) {
    const float* x    = (const float*)d_in[0];
    const int*   ei   = (const int*)  d_in[1];
    const float* q    = (const float*)d_in[2];
    const int*   bidx = (const int*)  d_in[3];
    const float* Win  = (const float*)d_in[4];
    const float* bin  = (const float*)d_in[5];
    const float* Wl[3] = {(const float*)d_in[6],  (const float*)d_in[10], (const float*)d_in[14]};
    const float* As[3] = {(const float*)d_in[7],  (const float*)d_in[11], (const float*)d_in[15]};
    const float* Ad[3] = {(const float*)d_in[8],  (const float*)d_in[12], (const float*)d_in[16]};
    const float* Bl[3] = {(const float*)d_in[9],  (const float*)d_in[13], (const float*)d_in[17]};
    const float* Wc   = (const float*)d_in[18];
    const float* bc   = (const float*)d_in[19];
    float* out = (float*)d_out;

    // ---- workspace carve ----
    char* p = (char*)d_ws;
    auto take = [&](size_t bytes) -> void* {
        void* r = (void*)p;
        p += (bytes + 255) & ~(size_t)255;
        return r;
    };
    unsigned short* comb = (unsigned short*)take((size_t)NN * KP * 2);
    unsigned short* winb = (unsigned short*)take((size_t)HID * KP * 2);
    unsigned short* wlb0 = (unsigned short*)take((size_t)HC * HID * 2);
    unsigned short* wlb1 = (unsigned short*)take((size_t)HC * HID * 2);
    unsigned short* wlb2 = (unsigned short*)take((size_t)HC * HID * 2);
    unsigned short* wlb[3] = {wlb0, wlb1, wlb2};
    unsigned short* Hb  = (unsigned short*)take((size_t)NN * HID * 2);
    float*    Hf   = (float*)take((size_t)NN * HID * 4);
    float*    XH   = (float*)take((size_t)NN * HC * 4);
    float*    oacc = (float*)take((size_t)NN * HC * 4);
    float*    als  = (float*)take((size_t)NN * HD * 4);
    float*    ald  = (float*)take((size_t)NN * HD * 4);
    unsigned* mmax = (unsigned*)take((size_t)NN * HD * 4);
    float*    ssum = (float*)take((size_t)NN * HD * 4);
    float*    elog = (float*)take((size_t)ET * HD * 4);

    const int T = 256;
    auto blocks = [&](long long n) { return (int)((n + T - 1) / T); };

    // ---- prep ----
    k_build_comb<<<blocks((long long)NN * KP), T, 0, stream>>>(x, q, bidx, comb);
    k_conv_win<<<blocks(HID * KP), T, 0, stream>>>(Win, winb);
    for (int l = 0; l < 3; ++l)
        k_conv_wl<<<blocks(HC * HID), T, 0, stream>>>(Wl[l], wlb[l], HC * HID);

    // ---- input projection (WMMA bf16) ----
    k_proj_gemm<<<NN / (16 * MT), 256, 0, stream>>>(comb, winb, bin, Hb);

    // ---- 3 GAT layers ----
    for (int l = 0; l < 3; ++l) {
        k_layer_gemm<<<dim3(NN / (16 * MT), HC / 128), 256, 0, stream>>>(Hb, wlb[l], XH);
        k_attn_logits<<<blocks(NN * HD), T, 0, stream>>>(XH, As[l], Ad[l], als, ald);
        k_init_ms<<<blocks(NN * HD), T, 0, stream>>>(mmax, ssum);
        k_zero4<<<blocks((long long)NN * HC / 4), T, 0, stream>>>((float4*)oacc,
                                                                  (long long)NN * HC / 4);
        k_edge_max<<<blocks((long long)ET * HD), T, 0, stream>>>(ei, als, ald, elog, mmax);
        k_edge_exp<<<blocks((long long)ET * HD), T, 0, stream>>>(ei, elog, mmax, ssum);
        k_edge_scatter<<<blocks((long long)ET * HID), T, 0, stream>>>(ei, elog, ssum, XH, oacc);
        k_reduce<<<blocks(NN * HID), T, 0, stream>>>(oacc, Bl[l], Hf, Hb);
    }

    // ---- classifier ----
    k_logits<<<blocks(NN), T, 0, stream>>>(Hf, Wc, bc, out);
}
